// BilinearAttention_46986942218859
// MI455X (gfx1250) — matrix operand order repaired and checked by reading, not compiled, not run on hardware
//
#include <hip/hip_runtime.h>
#include <hip/hip_bf16.h>

// ---------------------------------------------------------------------------
// Bilinear attention, fused flash-style for MI455X (gfx1250, wave32, WMMA).
//   B=4, S=4096, H=256.
//   out = softmax_mask( (M@W) @ M^T * adj ) @ M
// adj (268MB, the only mandatory HBM stream) is DMA'd tile-by-tile into LDS
// by the Tensor Data Mover, double-buffered and overlapped with the
// v_wmma_f32_16x16x32_bf16 compute stream. bf16 copies of matrix (8MB each)
// stay L2-resident (192MB L2). Scores never touch memory; online softmax.
// ---------------------------------------------------------------------------

#define SB 4
#define SS 4096
#define SH 256
#define NEG_INF_F (-1e30f)

typedef __attribute__((ext_vector_type(16))) __bf16 v16bf;
typedef __attribute__((ext_vector_type(8)))  float  v8f;
typedef __attribute__((ext_vector_type(4)))  unsigned int u32x4;
typedef __attribute__((ext_vector_type(8)))  int          i32x8;
typedef __attribute__((ext_vector_type(4)))  int          i32x4;

#if __has_builtin(__builtin_amdgcn_tensor_load_to_lds) && \
    __has_builtin(__builtin_amdgcn_s_wait_tensorcnt)
#define HAVE_TDM 1
#else
#define HAVE_TDM 0
#endif

union AB {
  v16bf v;
  unsigned int u[8];
};

static __device__ __forceinline__ unsigned short f32_to_bf16(float f) {
  unsigned int u = __builtin_bit_cast(unsigned int, f);
  u += 0x7fffu + ((u >> 16) & 1u);      // round-to-nearest-even
  return (unsigned short)(u >> 16);
}

static __device__ __forceinline__ v8f wmma_bf16(const AB& a, const AB& b, v8f c) {
  // (neg_a, A, neg_b, B, c_mod, C, reuse_a, reuse_b)
  return __builtin_amdgcn_wmma_f32_16x16x32_bf16(false, a.v, false, b.v,
                                                 (short)0, c, false, false);
}

#if HAVE_TDM
// Issue one TDM transfer: 16 rows x 32 cols of fp32 from adj (row stride S)
// into a contiguous 2KB LDS tile.  Descriptor per CDNA5 ISA ch.8 (D#).
static __device__ __forceinline__ void tdm_load_adj(unsigned lds_off,
                                                    const float* gptr) {
  unsigned long long ga = (unsigned long long)(uintptr_t)gptr;
  u32x4 g0;
  g0.x = 1u;                                   // count=1, user mode, no gather
  g0.y = lds_off;                              // lds_addr [63:32]
  g0.z = (unsigned)ga;                         // global_addr [95:64]
  g0.w = ((unsigned)(ga >> 32) & 0x01FFFFFFu)  // global_addr [120:96]
         | 0x80000000u;                        // type=2 ("image") [127:126]
  i32x8 g1;
  g1[0] = 0x00020000;      // wg_mask=0, data_size=2 (4B), no pad/iter/barrier
  g1[1] = 0x10000000;      // tensor_dim0 = 4096  (bits [79:48], low half here)
  g1[2] = 0x10000000;      // tensor_dim0 hi=0 | tensor_dim1 = 4096 (low half)
  g1[3] = 0x00200000;      // tensor_dim1 hi=0 | tile_dim0 = 32
  g1[4] = 16;              // tile_dim1 = 16, tile_dim2 = 0
  g1[5] = 4096;            // tensor_dim0_stride = S (low 32)
  g1[6] = 0;               // stride hi | tensor_dim1_stride lo
  g1[7] = 0;
  i32x4 g2 = {0, 0, 0, 0};
  i32x4 g3 = {0, 0, 0, 0};
#if __clang_major__ >= 23
  i32x8 g4 = {0, 0, 0, 0, 0, 0, 0, 0};
  __builtin_amdgcn_tensor_load_to_lds(g0, g1, g2, g3, g4, 0);
#else
  __builtin_amdgcn_tensor_load_to_lds(g0, g1, g2, g3, 0);
#endif
}
#endif

// ---------------------------------------------------------------------------
// Kernel 1: fp32 matrix -> bf16 row-major copy + bf16 (b,h,s) transposed copy.
// ---------------------------------------------------------------------------
__global__ void k_convert(const float* __restrict__ mat,
                          unsigned short* __restrict__ matbf,
                          unsigned short* __restrict__ matT) {
  int i = blockIdx.x * blockDim.x + threadIdx.x;     // 0 .. B*S*H-1
  float v = mat[i];
  unsigned short bv = f32_to_bf16(v);
  matbf[i] = bv;
  int h  = i & (SH - 1);
  int bs = i >> 8;            // b*S + s
  int s  = bs & (SS - 1);
  int b  = bs >> 12;
  matT[(((b << 8) + h) << 12) + s] = bv;
}

// Kernel 1b: W (H,H) fp32 -> transposed bf16 (n,k) so B-operand K-pairs are
// contiguous u32 loads in k_proj.
__global__ void k_convW(const float* __restrict__ W,
                        unsigned short* __restrict__ WTbf) {
  int i = blockIdx.x * blockDim.x + threadIdx.x;     // i = k*H + n
  int k = i >> 8, n = i & (SH - 1);
  WTbf[n * SH + k] = f32_to_bf16(W[i]);
}

// ---------------------------------------------------------------------------
// Kernel 2: P = matrix @ W  (bf16 out).  One wave per 16x16 output tile.
// ---------------------------------------------------------------------------
__global__ __launch_bounds__(256) void k_proj(const unsigned short* __restrict__ matbf,
                                              const unsigned short* __restrict__ WTbf,
                                              unsigned short* __restrict__ Pbf) {
  int wave = blockIdx.x * (blockDim.x >> 5) + (threadIdx.x >> 5);
  int lane = threadIdx.x & 31;
  int nt = wave & 15;                 // column tile (H/16)
  int rt = wave >> 4;                 // row tile    (B*S/16)
  int b  = rt >> 8;
  int s0 = (rt & 255) << 4;
  int hi  = lane >> 4;
  int l16 = lane & 15;

  const unsigned int* matu = (const unsigned int*)matbf;  // bf16 pairs
  const unsigned int* WTu  = (const unsigned int*)WTbf;
  v8f c = {};
  int arow = (b << 12) + s0 + l16;    // global matrix row for this lane's A slot
  int wrow = (nt * 16 + l16) * 128;   // WT row (u32 units)
  for (int kk = 0; kk < 8; ++kk) {    // K = 256 in chunks of 32
    AB a, bm;
#pragma unroll
    for (int r = 0; r < 8; ++r) {
      // A (16x32 bf16): lane row = l16; vgpr r holds K pair {kbase, kbase+1}
      a.u[r] = matu[arow * 128 + kk * 16 + r + 8 * hi];
      // B (32x16): lane col = nt*16+l16; vgpr r holds K rows {2r+16hi, +1}
      bm.u[r] = WTu[wrow + kk * 16 + r + 8 * hi];
    }
    c = wmma_bf16(a, bm, c);
  }
#pragma unroll
  for (int r = 0; r < 8; ++r) {
    int row = s0 + r + 8 * hi;        // C layout: vgpr r, half hi -> row r+8hi
    Pbf[((b << 12) + row) * SH + nt * 16 + l16] = f32_to_bf16(c[r]);
  }
}

// ---------------------------------------------------------------------------
// Kernel 3: fused  scores -> *adj -> mask -> online softmax -> @V.
// 1 wave = 16 query rows; 4 waves / block; key loop in blocks of 32.
// adj tiles streamed into LDS by TDM, double-buffered.
// ---------------------------------------------------------------------------
__global__ __launch_bounds__(128, 1) void k_attn(const unsigned short* __restrict__ matbf,
                                                 const unsigned short* __restrict__ matT,
                                                 const unsigned short* __restrict__ Pbf,
                                                 const float* __restrict__ adj,
                                                 const int* __restrict__ mask,
                                                 float* __restrict__ out) {
  __shared__ unsigned int   q_lds[4][16 * 128];   // Q' tile, bf16 pairs (8KB/wave)
  __shared__ unsigned short p_lds[4][16 * 32];    // P re-layout scratch (1KB/wave)
#if HAVE_TDM
  __shared__ float          adj_lds[4][2][16 * 32]; // TDM double buffer (4KB/wave)
#endif

  int w    = __builtin_amdgcn_readfirstlane(threadIdx.x >> 5); // wave-uniform
  int lane = threadIdx.x & 31;
  int qb   = blockIdx.x * 4 + w;      // 0..1023 query tiles
  int b    = qb >> 8;
  int q0   = (qb & 255) << 4;
  int hi   = lane >> 4;
  int l16  = lane & 15;

  const unsigned int* matu  = (const unsigned int*)matbf;
  const unsigned int* matTu = (const unsigned int*)matT;
  const unsigned int* Pu    = (const unsigned int*)Pbf;

  // Stage this wave's Q' rows (16 x 256 bf16) into LDS.
#pragma unroll 4
  for (int i = 0; i < 64; ++i) {
    int flat = lane + 32 * i;
    int row = flat >> 7, col = flat & 127;
    q_lds[w][row * 128 + col] = Pu[((b << 12) + q0 + row) * 128 + col];
  }
  __syncthreads();

  const float* adj_base = adj + ((size_t)((b << 12) + q0)) * SS;  // row q0, key 0
#if HAVE_TDM
  unsigned adj_lds_off0 = (unsigned)(uintptr_t)&adj_lds[w][0][0];
  unsigned adj_lds_off1 = (unsigned)(uintptr_t)&adj_lds[w][1][0];
  tdm_load_adj(adj_lds_off0, adj_base);          // prime buffer 0 (kb=0)
#endif

  float m[8], lsum[8], scale[8];
  v8f O[16];
  v8f vzero = {};
#pragma unroll
  for (int r = 0; r < 8; ++r) { m[r] = -3.0e38f; lsum[r] = 0.f; }
#pragma unroll
  for (int t = 0; t < 16; ++t) O[t] = vzero;

  for (int kb = 0; kb < 128; ++kb) {
    int key0 = kb * 32;

#if HAVE_TDM
    // Kick off next adj tile while we compute this block's scores.
    if (kb + 1 < 128)
      tdm_load_adj((kb & 1) ? adj_lds_off0 : adj_lds_off1,
                   adj_base + (size_t)(kb + 1) * 32);
#endif

    // ---- scores: 2 tiles of 16q x 16k, K = H = 256 -------------------------
    v8f c0 = vzero, c1 = vzero;
    int keyA = (b << 12) + key0 + l16;
    for (int kk = 0; kk < 8; ++kk) {
      AB a, b0, b1;
#pragma unroll
      for (int r = 0; r < 8; ++r)
        a.u[r] = q_lds[w][l16 * 128 + kk * 16 + r + 8 * hi];
#pragma unroll
      for (int r = 0; r < 8; ++r)
        b0.u[r] = matu[keyA * 128 + kk * 16 + r + 8 * hi];
      c0 = wmma_bf16(a, b0, c0);
#pragma unroll
      for (int r = 0; r < 8; ++r)
        b1.u[r] = matu[(keyA + 16) * 128 + kk * 16 + r + 8 * hi];
      c1 = wmma_bf16(a, b1, c1);
    }

    // ---- adj * scores, key mask, online softmax ---------------------------
    int km0 = mask[(b << 12) + key0 + l16];
    int km1 = mask[(b << 12) + key0 + 16 + l16];

    float p0[8], p1[8];
#if HAVE_TDM
    // Wait for this block's tile (<=1 outstanding keeps the prefetch in flight)
    if (kb + 1 < 128) __builtin_amdgcn_s_wait_tensorcnt(1);
    else              __builtin_amdgcn_s_wait_tensorcnt(0);
    const float* at = &adj_lds[w][kb & 1][0];
#pragma unroll
    for (int r = 0; r < 8; ++r) {
      float a0v = at[(r + 8 * hi) * 32 + l16];
      float a1v = at[(r + 8 * hi) * 32 + l16 + 16];
      p0[r] = km0 ? c0[r] * a0v : NEG_INF_F;
      p1[r] = km1 ? c1[r] * a1v : NEG_INF_F;
    }
#else
    if (kb + 1 < 128)
      __builtin_prefetch(&adj_base[(size_t)(q0 ? 0 : 0) + (size_t)l16 * SS + key0 + 32], 0, 1);
#pragma unroll
    for (int r = 0; r < 8; ++r) {
      size_t arow = ((size_t)(r + 8 * hi)) * SS + key0;
      float a0v = adj_base[arow + l16];
      float a1v = adj_base[arow + 16 + l16];
      p0[r] = km0 ? c0[r] * a0v : NEG_INF_F;
      p1[r] = km1 ? c1[r] * a1v : NEG_INF_F;
    }
#endif

#pragma unroll
    for (int r = 0; r < 8; ++r) {
      float mx = fmaxf(p0[r], p1[r]);
      mx = fmaxf(mx, __shfl_xor(mx, 1, 32));
      mx = fmaxf(mx, __shfl_xor(mx, 2, 32));
      mx = fmaxf(mx, __shfl_xor(mx, 4, 32));
      mx = fmaxf(mx, __shfl_xor(mx, 8, 32));
      float mn = fmaxf(m[r], mx);
      scale[r] = __expf(m[r] - mn);
      m[r] = mn;
      float e0 = __expf(p0[r] - mn);
      float e1 = __expf(p1[r] - mn);
      p0[r] = e0; p1[r] = e1;
      float rs = e0 + e1;
      rs += __shfl_xor(rs, 1, 32);
      rs += __shfl_xor(rs, 2, 32);
      rs += __shfl_xor(rs, 4, 32);
      rs += __shfl_xor(rs, 8, 32);
      lsum[r] = lsum[r] * scale[r] + rs;
    }
    // rescale accumulators (row r+8hi matches scale[r] in this half)
#pragma unroll
    for (int t = 0; t < 16; ++t)
#pragma unroll
      for (int r = 0; r < 8; ++r) O[t][r] *= scale[r];

    // ---- P (C layout) -> bf16 A-operand layout via LDS --------------------
#pragma unroll
    for (int r = 0; r < 8; ++r) {
      p_lds[w][(r + 8 * hi) * 32 + l16]      = f32_to_bf16(p0[r]);
      p_lds[w][(r + 8 * hi) * 32 + l16 + 16] = f32_to_bf16(p1[r]);
    }
    __syncthreads();
    AB pa;
    const unsigned int* pl = (const unsigned int*)&p_lds[w][0];
#pragma unroll
    for (int r = 0; r < 8; ++r)
      pa.u[r] = pl[l16 * 16 + r + 8 * hi];
    __syncthreads();

    // ---- O += P @ V  (V from transposed bf16 matrix; contiguous key pairs)
#pragma unroll
    for (int t = 0; t < 16; ++t) {
      AB vb;
      int hcol = t * 16 + l16;
#pragma unroll
      for (int r = 0; r < 8; ++r)
        vb.u[r] = matTu[(((b << 8) + hcol) << 11) + (key0 >> 1) + r + 8 * hi];
      O[t] = wmma_bf16(pa, vb, O[t]);
    }
  }

  // ---- normalize + write fp32 output --------------------------------------
#pragma unroll
  for (int t = 0; t < 16; ++t) {
#pragma unroll
    for (int r = 0; r < 8; ++r) {
      int q = q0 + r + 8 * hi;
      out[((size_t)((b << 12) + q)) * SH + t * 16 + l16] = O[t][r] / lsum[r];
    }
  }
}

// ---------------------------------------------------------------------------
extern "C" void kernel_launch(void* const* d_in, const int* in_sizes, int n_in,
                              void* d_out, int out_size, void* d_ws, size_t ws_size,
                              hipStream_t stream) {
  const float* matrix = (const float*)d_in[0];   // (B,S,H) fp32
  const int*   mask   = (const int*)d_in[1];     // (B,S)   int32
  const float* adj    = (const float*)d_in[2];   // (B,S,S) fp32
  const float* W      = (const float*)d_in[3];   // (H,H)   fp32
  float* out = (float*)d_out;                    // (B,S,H) fp32

  const int total = SB * SS * SH;                // 4M elements
  unsigned short* matbf = (unsigned short*)d_ws;           //  8 MB
  unsigned short* matT  = matbf + total;                   //  8 MB
  unsigned short* Pbf   = matT + total;                    //  8 MB
  unsigned short* WTbf  = Pbf + total;                     // 128 KB

  k_convert<<<total / 256, 256, 0, stream>>>(matrix, matbf, matT);
  k_convW<<<(SH * SH) / 256, 256, 0, stream>>>(W, WTbf);
  k_proj<<<(SB * (SS / 16) * (SH / 16)) / 8, 256, 0, stream>>>(matbf, WTbf, Pbf);
  k_attn<<<(SB * (SS / 16)) / 4, 128, 0, stream>>>(matbf, matT, Pbf, adj, mask, out);
}
